// GeneralConv_30820685316781
// MI455X (gfx1250) — matrix operand order repaired and checked by reading, not compiled, not run on hardware
//
#include <hip/hip_runtime.h>
#include <hip/hip_bf16.h>
#include <math.h>

// ---------------------------------------------------------------- constants
constexpr int NN = 40000;
constexpr int EE = 320000;
constexpr int DD = 128;
constexpr int HH = 4;
constexpr int TT = 3;
constexpr int RR = 5;
constexpr int KM = 3;          // moments K
constexpr int DK = 32;
#define RSQRT_DK 0.17677669529663689f   // 1/sqrt(32)

typedef __attribute__((ext_vector_type(16))) __bf16 v16bf;
typedef __attribute__((ext_vector_type(8)))  float  v8f;

union BFPack { v16bf v; uint4 q[2]; };

// ---------------------------------------------------------------- WMMA utils
__device__ __forceinline__ v8f wmma_bf16(v16bf a, v16bf b, v8f c) {
    return __builtin_amdgcn_wmma_f32_16x16x32_bf16(
        false, a, false, b, (short)0, c, false, false);
}

// A fragment: 16x32 bf16, rows gathered via rows[] (LDS), row-major, stride ld.
// Layout (ISA 7.12.2): lane<16 -> row=lane, elems {K0..7, K16..23};
//                      lane>=16 -> row=lane-16, elems {K8..15, K24..31}.
__device__ __forceinline__ v16bf load_a_frag(const __bf16* base, const int* rows,
                                             int ld, int k0) {
    int lane = threadIdx.x & 31;
    int half = lane >> 4;
    int m    = lane & 15;
    const __bf16* rp = base + (size_t)rows[m] * ld;
    BFPack u;
    u.q[0] = *(const uint4*)(rp + k0 + half * 8);
    u.q[1] = *(const uint4*)(rp + k0 + 16 + half * 8);
    return u.v;
}

// B fragment: 32x16 bf16 from transposed weight Wt[out][in] (row-major, stride ld).
// lane<16 -> col=lane, K0..7,K16..23 ; lane>=16 -> col=lane-16, K8..15,K24..31.
__device__ __forceinline__ v16bf load_b_frag(const __bf16* wt, int ld,
                                             int n0, int k0) {
    int lane = threadIdx.x & 31;
    int half = lane >> 4;
    int n    = lane & 15;
    const __bf16* cp = wt + (size_t)(n0 + n) * ld + k0 + half * 8;
    BFPack u;
    u.q[0] = *(const uint4*)(cp);
    u.q[1] = *(const uint4*)(cp + 16);
    return u.v;
}

// monotonic float ordering encode/decode for atomicMax-based segment max
__device__ __forceinline__ unsigned enc_f32(float f) {
    unsigned b = __float_as_uint(f);
    return (b & 0x80000000u) ? ~b : (b | 0x80000000u);
}
__device__ __forceinline__ float dec_f32(unsigned u) {
    unsigned b = (u & 0x80000000u) ? (u ^ 0x80000000u) : ~u;
    return __uint_as_float(b);
}

// ---------------------------------------------------------------- bookkeeping
__global__ void count_kernel(const int* __restrict__ ntype,
                             const int* __restrict__ etype,
                             const int* __restrict__ tgt,
                             int* tcnt, int* rcnt, int* deg) {
    int i = blockIdx.x * blockDim.x + threadIdx.x;
    if (i < NN) atomicAdd(&tcnt[ntype[i]], 1);
    if (i < EE) { atomicAdd(&rcnt[etype[i]], 1); atomicAdd(&deg[tgt[i]], 1); }
}

__global__ void scan_kernel(const int* tcnt, const int* rcnt,
                            int* toff, int* roff, int* tcur, int* rcur) {
    if (blockIdx.x == 0 && threadIdx.x == 0) {
        int a = 0;
        for (int t = 0; t < TT; ++t) { toff[t] = a; tcur[t] = a; a += tcnt[t]; }
        int b = 0;
        for (int r = 0; r < RR; ++r) { roff[r] = b; rcur[r] = b; b += rcnt[r]; }
    }
}

__global__ void scatter_kernel(const int* __restrict__ ntype,
                               const int* __restrict__ etype,
                               int* tcur, int* rcur,
                               int* nsorted, int* esorted) {
    int i = blockIdx.x * blockDim.x + threadIdx.x;
    if (i < NN) { int p = atomicAdd(&tcur[ntype[i]], 1); nsorted[p] = i; }
    if (i < EE) { int p = atomicAdd(&rcur[etype[i]], 1); esorted[p] = i; }
}

// ---------------------------------------------------------------- casts
__global__ void cast_x_kernel(const float* __restrict__ x, __bf16* __restrict__ xb,
                              int n) {
    int i = blockIdx.x * blockDim.x + threadIdx.x;
    if (i < n) xb[i] = (__bf16)x[i];
}

// transpose-cast: src [nmat][dim][dim] (in-major) -> dst [nmat][dim][dim] (out-major)
__global__ void castT_kernel(const float* __restrict__ src, __bf16* __restrict__ dst,
                             int dim, int nmat) {
    int i = blockIdx.x * blockDim.x + threadIdx.x;
    int tot = nmat * dim * dim;
    if (i >= tot) return;
    int g = i / (dim * dim);
    int r = i - g * dim * dim;
    int o = r / dim, in = r - o * dim;
    dst[i] = (__bf16)src[(size_t)g * dim * dim + (size_t)in * dim + o];
}

// ---------------------------------------------------------------- typed q/k/v
__global__ void typed_qkv_kernel(const __bf16* __restrict__ xbf,
                                 const int* __restrict__ nsorted,
                                 const int* __restrict__ tcnt, const int* __restrict__ toff,
                                 const __bf16* __restrict__ qwT, const __bf16* __restrict__ kwT,
                                 const __bf16* __restrict__ vwT,
                                 const float* __restrict__ qb, const float* __restrict__ kb,
                                 const float* __restrict__ vb,
                                 float* __restrict__ qn,
                                 __bf16* __restrict__ knbf, __bf16* __restrict__ vnbf) {
    __shared__ int rows[16];
    __shared__ int s_t, s_base, s_valid;
    if (threadIdx.x == 0) {
        int rem = blockIdx.x, t = 0;
        for (; t < TT; ++t) {
            int nt = (tcnt[t] + 15) >> 4;
            if (rem < nt) break;
            rem -= nt;
        }
        s_t = t;
        if (t < TT) {
            s_base = toff[t] + rem * 16;
            int v = tcnt[t] - rem * 16;
            s_valid = v > 16 ? 16 : v;
        }
    }
    __syncthreads();
    if (s_t >= TT) return;
    if (threadIdx.x < 16) {
        int j = (int)threadIdx.x;
        int cl = j < s_valid ? j : s_valid - 1;
        rows[j] = nsorted[s_base + cl];
    }
    __syncthreads();

    int t = s_t;
    int wave = threadIdx.x >> 5;
    int n0 = wave * 16;
    const __bf16* qw = qwT + (size_t)t * DD * DD;
    const __bf16* kw = kwT + (size_t)t * DD * DD;
    const __bf16* vw = vwT + (size_t)t * DD * DD;
    v8f aq = {}, ak = {}, av = {};
    for (int k0 = 0; k0 < DD; k0 += 32) {
        v16bf a = load_a_frag(xbf, rows, DD, k0);
        aq = wmma_bf16(a, load_b_frag(qw, DD, n0, k0), aq);
        ak = wmma_bf16(a, load_b_frag(kw, DD, n0, k0), ak);
        av = wmma_bf16(a, load_b_frag(vw, DD, n0, k0), av);
    }
    int lane = threadIdx.x & 31;
    int col  = n0 + (lane & 15);
    int mb   = (lane >> 4) * 8;
    float bq_ = qb[t * DD + col], bk_ = kb[t * DD + col], bv_ = vb[t * DD + col];
    for (int j = 0; j < 8; ++j) {
        int mi = mb + j;
        if (mi < s_valid) {
            int node = rows[mi];
            qn[(size_t)node * DD + col]   = aq[j] + bq_;
            knbf[(size_t)node * DD + col] = (__bf16)(ak[j] + bk_);
            vnbf[(size_t)node * DD + col] = (__bf16)(av[j] + bv_);
        }
    }
}

// ---------------------------------------------------------------- edge stage
__global__ void edge_attn_kernel(const __bf16* __restrict__ knbf,
                                 const __bf16* __restrict__ vnbf,
                                 const float* __restrict__ qn,
                                 const int* __restrict__ esorted,
                                 const int* __restrict__ rcnt, const int* __restrict__ roff,
                                 const int* __restrict__ src, const int* __restrict__ tgt,
                                 const __bf16* __restrict__ rattT,
                                 const __bf16* __restrict__ rmsgT,
                                 const float* __restrict__ rel_pri,
                                 float* __restrict__ lg, unsigned* __restrict__ mxu,
                                 float* __restrict__ vps) {
    __shared__ int eids[16], srows[16];
    __shared__ float kp[16][DD];
    __shared__ int s_r, s_base, s_valid;
    if (threadIdx.x == 0) {
        int rem = blockIdx.x, r = 0;
        for (; r < RR; ++r) {
            int nt = (rcnt[r] + 15) >> 4;
            if (rem < nt) break;
            rem -= nt;
        }
        s_r = r;
        if (r < RR) {
            s_base = roff[r] + rem * 16;
            int v = rcnt[r] - rem * 16;
            s_valid = v > 16 ? 16 : v;
        }
    }
    __syncthreads();
    if (s_r >= RR) return;
    if (threadIdx.x < 16) {
        int j = (int)threadIdx.x;
        int cl = j < s_valid ? j : s_valid - 1;
        int e = esorted[s_base + cl];
        eids[j] = e;
        srows[j] = src[e];
    }
    __syncthreads();

    int r = s_r;
    int wave  = threadIdx.x >> 5;
    int h     = wave >> 1;       // head 0..3
    int halfc = wave & 1;        // output-column half
    int n0 = halfc * 16;
    const __bf16* batt = rattT + (size_t)(r * HH + h) * DK * DK;
    const __bf16* bmsg = rmsgT + (size_t)(r * HH + h) * DK * DK;
    v8f zc = {};
    v16bf akf = load_a_frag(knbf, srows, DD, h * DK);
    v8f ck = wmma_bf16(akf, load_b_frag(batt, DK, n0, 0), zc);
    v16bf avf = load_a_frag(vnbf, srows, DD, h * DK);
    v8f cv = wmma_bf16(avf, load_b_frag(bmsg, DK, n0, 0), zc);

    int lane = threadIdx.x & 31;
    int col  = h * DK + n0 + (lane & 15);
    int mb   = (lane >> 4) * 8;
    for (int j = 0; j < 8; ++j) {
        int mi = mb + j;
        kp[mi][col] = ck[j];
        if (mi < s_valid) {
            int e = eids[mi];
            if (e < NN) vps[(size_t)e * DD + col] = cv[j];
        }
    }
    __syncthreads();

    if (threadIdx.x < 64) {
        int mi = threadIdx.x >> 2;
        int hh = threadIdx.x & 3;
        if (mi < s_valid) {
            int e = eids[mi];
            int tn = tgt[e];
            const float* q = qn + (size_t)tn * DD + hh * DK;
            float s = 0.f;
            #pragma unroll
            for (int d = 0; d < DK; ++d) s += q[d] * kp[mi][hh * DK + d];
            float lgt = s * rel_pri[r * HH + hh] * RSQRT_DK;
            lg[(size_t)e * HH + hh] = lgt;
            atomicMax(&mxu[(size_t)tn * HH + hh], enc_f32(lgt));
        }
    }
}

__global__ void softmax_ex_kernel(const int* __restrict__ tgt,
                                  float* __restrict__ lg,
                                  const unsigned* __restrict__ mxu,
                                  float* __restrict__ ssum) {
    int i = blockIdx.x * blockDim.x + threadIdx.x;
    if (i >= EE * HH) return;
    int e = i >> 2, h = i & 3;
    int t = tgt[e];
    float ex = __expf(lg[i] - dec_f32(mxu[(size_t)t * HH + h]));
    lg[i] = ex;
    atomicAdd(&ssum[(size_t)t * HH + h], ex);
}

// agg[k][n] = deg[n] * vp[edge n]^(k+1) * att[edge n]   (faithful tgt-index quirk)
__global__ void moments_kernel(const float* __restrict__ vps,
                               const float* __restrict__ lg,
                               const float* __restrict__ ssum,
                               const int* __restrict__ deg,
                               const int* __restrict__ tgt,
                               __bf16* __restrict__ aggbf) {
    int i = blockIdx.x * blockDim.x + threadIdx.x;
    if (i >= NN * DD) return;
    int n = i >> 7, d = i & 127, h = d >> 5;
    int t = tgt[n];  // target of EDGE id n
    float att = lg[(size_t)n * HH + h] / (ssum[(size_t)t * HH + h] + 1e-16f);
    float v = vps[i];
    float dg = (float)deg[n];
    float m1 = v * att * dg;
    float m2 = v * v * att * dg;
    float m3 = v * v * v * att * dg;
    float sg = (m3 > 0.f) ? 1.f : ((m3 < 0.f) ? -1.f : 0.f);
    m3 = sg * powf(fabsf(m3) + 1e-18f, 1.0f / 3.0f);
    aggbf[i] = (__bf16)m1;
    aggbf[(size_t)NN * DD + i] = (__bf16)m2;
    aggbf[(size_t)2 * NN * DD + i] = (__bf16)m3;
}

// ---------------------------------------------------------------- generic GEMM
// C[M x 128] = A(bf16, row-major) @ Bt(bf16 [out][in]) (+bias); C / Cbf nullable
__global__ void gemm128_kernel(const __bf16* __restrict__ A,
                               const __bf16* __restrict__ Bt,
                               const float* __restrict__ bias,
                               float* __restrict__ C, __bf16* __restrict__ Cbf,
                               int M) {
    __shared__ int rows[16];
    int r0 = blockIdx.x * 16;
    if (threadIdx.x < 16) {
        int rr = r0 + (int)threadIdx.x;
        rows[threadIdx.x] = rr < M ? rr : M - 1;
    }
    __syncthreads();
    int wave = threadIdx.x >> 5;
    int n0 = wave * 16;
    v8f acc = {};
    for (int k0 = 0; k0 < DD; k0 += 32) {
        v16bf a = load_a_frag(A, rows, DD, k0);
        acc = wmma_bf16(a, load_b_frag(Bt, DD, n0, k0), acc);
    }
    int lane = threadIdx.x & 31;
    int col  = n0 + (lane & 15);
    int mb   = (lane >> 4) * 8;
    float bv = bias ? bias[col] : 0.f;
    for (int j = 0; j < 8; ++j) {
        int m = r0 + mb + j;
        if (m < M) {
            float v = acc[j] + bv;
            if (C)   C[(size_t)m * DD + col] = v;
            if (Cbf) Cbf[(size_t)m * DD + col] = (__bf16)v;
        }
    }
}

// res += sigmoid(rowdot(front, tail)) * aggm_k
__global__ void gate_kernel(const float* __restrict__ front,
                            const float* __restrict__ tail,
                            const float* __restrict__ aggm_k,
                            float* __restrict__ res) {
    __shared__ float red[4];
    int n = blockIdx.x, d = threadIdx.x;
    float p = front[(size_t)n * DD + d] * tail[(size_t)n * DD + d];
    for (int o = 16; o; o >>= 1) p += __shfl_down(p, o, 32);
    if ((d & 31) == 0) red[d >> 5] = p;
    __syncthreads();
    float s = red[0] + red[1] + red[2] + red[3];
    float t = 1.f / (1.f + __expf(-s));
    res[(size_t)n * DD + d] += t * aggm_k[(size_t)n * DD + d];
}

__global__ void gelu_cast_kernel(const float* __restrict__ res,
                                 __bf16* __restrict__ hbf, int n) {
    int i = blockIdx.x * blockDim.x + threadIdx.x;
    if (i >= n) return;
    float x = res[i];
    hbf[i] = (__bf16)(0.5f * x * (1.f + erff(x * 0.70710678118654752f)));
}

// ---------------------------------------------------------------- typed output
__global__ void typed_out_kernel(const __bf16* __restrict__ hbf,
                                 const int* __restrict__ nsorted,
                                 const int* __restrict__ tcnt, const int* __restrict__ toff,
                                 const __bf16* __restrict__ awT,
                                 const float* __restrict__ ab,
                                 float* __restrict__ trans) {
    __shared__ int rows[16];
    __shared__ int s_t, s_base, s_valid;
    if (threadIdx.x == 0) {
        int rem = blockIdx.x, t = 0;
        for (; t < TT; ++t) {
            int nt = (tcnt[t] + 15) >> 4;
            if (rem < nt) break;
            rem -= nt;
        }
        s_t = t;
        if (t < TT) {
            s_base = toff[t] + rem * 16;
            int v = tcnt[t] - rem * 16;
            s_valid = v > 16 ? 16 : v;
        }
    }
    __syncthreads();
    if (s_t >= TT) return;
    if (threadIdx.x < 16) {
        int j = (int)threadIdx.x;
        int cl = j < s_valid ? j : s_valid - 1;
        rows[j] = nsorted[s_base + cl];
    }
    __syncthreads();
    int t = s_t;
    int wave = threadIdx.x >> 5;
    int n0 = wave * 16;
    const __bf16* aw = awT + (size_t)t * DD * DD;
    v8f acc = {};
    for (int k0 = 0; k0 < DD; k0 += 32) {
        v16bf a = load_a_frag(hbf, rows, DD, k0);
        acc = wmma_bf16(a, load_b_frag(aw, DD, n0, k0), acc);
    }
    int lane = threadIdx.x & 31;
    int col  = n0 + (lane & 15);
    int mb   = (lane >> 4) * 8;
    float b = ab[t * DD + col];
    for (int j = 0; j < 8; ++j) {
        int mi = mb + j;
        if (mi < s_valid)
            trans[(size_t)rows[mi] * DD + col] = acc[j] + b;
    }
}

// ---------------------------------------------------------------- final LN
__global__ void final_ln_kernel(const float* __restrict__ trans,
                                const float* __restrict__ x,
                                const int* __restrict__ ntype,
                                const float* __restrict__ skip,
                                const float* __restrict__ ln_g,
                                const float* __restrict__ ln_b,
                                float* __restrict__ out) {
    __shared__ float sm[4], sv[4];
    int n = blockIdx.x, d = threadIdx.x;
    int t = ntype[n];
    float alpha = 1.f / (1.f + __expf(-skip[t]));
    float y = trans[(size_t)n * DD + d] * alpha + x[(size_t)n * DD + d] * (1.f - alpha);
    float s = y;
    for (int o = 16; o; o >>= 1) s += __shfl_down(s, o, 32);
    if ((d & 31) == 0) sm[d >> 5] = s;
    __syncthreads();
    float mu = (sm[0] + sm[1] + sm[2] + sm[3]) * (1.f / DD);
    float c = y - mu;
    float v2 = c * c;
    for (int o = 16; o; o >>= 1) v2 += __shfl_down(v2, o, 32);
    if ((d & 31) == 0) sv[d >> 5] = v2;
    __syncthreads();
    float var = (sv[0] + sv[1] + sv[2] + sv[3]) * (1.f / DD);
    out[(size_t)n * DD + d] = c * rsqrtf(var + 1e-5f) * ln_g[t * DD + d] + ln_b[t * DD + d];
}

// ---------------------------------------------------------------- host driver
extern "C" void kernel_launch(void* const* d_in, const int* in_sizes, int n_in,
                              void* d_out, int out_size, void* d_ws, size_t ws_size,
                              hipStream_t stream) {
    const float* meta_xs = (const float*)d_in[0];
    const int*   ntype   = (const int*)d_in[1];
    const int*   eidx    = (const int*)d_in[2];
    const int*   etype   = (const int*)d_in[3];
    const float* q_w = (const float*)d_in[5],  *q_b = (const float*)d_in[6];
    const float* k_w = (const float*)d_in[7],  *k_b = (const float*)d_in[8];
    const float* v_w = (const float*)d_in[9],  *v_b = (const float*)d_in[10];
    const float* a_w = (const float*)d_in[11], *a_b = (const float*)d_in[12];
    const float* rel_pri = (const float*)d_in[13];
    const float* rel_att = (const float*)d_in[14];
    const float* rel_msg = (const float*)d_in[15];
    const float* WMk = (const float*)d_in[16];
    const float* Wak = (const float*)d_in[17];
    const float* Wq  = (const float*)d_in[18], *bq  = (const float*)d_in[19];
    const float* Wkl = (const float*)d_in[20], *bkl = (const float*)d_in[21];
    const float* skip = (const float*)d_in[22];
    const float* ln_g = (const float*)d_in[23];
    const float* ln_b = (const float*)d_in[24];
    const int* src = eidx;
    const int* tgt = eidx + EE;
    float* out = (float*)d_out;

    // workspace carve-up (256B aligned)
    char* base = (char*)d_ws;
    size_t off = 0;
    auto alloc = [&](size_t bytes) -> void* {
        off = (off + 255) & ~(size_t)255;
        void* p = base + off;
        off += bytes;
        return p;
    };
    __bf16* xbf   = (__bf16*)alloc((size_t)NN * DD * 2);
    __bf16* qwT   = (__bf16*)alloc((size_t)TT * DD * DD * 2);
    __bf16* kwT   = (__bf16*)alloc((size_t)TT * DD * DD * 2);
    __bf16* vwT   = (__bf16*)alloc((size_t)TT * DD * DD * 2);
    __bf16* awT   = (__bf16*)alloc((size_t)TT * DD * DD * 2);
    __bf16* WMkT  = (__bf16*)alloc((size_t)KM * DD * DD * 2);
    __bf16* WakT  = (__bf16*)alloc((size_t)KM * DD * DD * 2);
    __bf16* WqT   = (__bf16*)alloc((size_t)DD * DD * 2);
    __bf16* WklT  = (__bf16*)alloc((size_t)DD * DD * 2);
    __bf16* rattT = (__bf16*)alloc((size_t)RR * HH * DK * DK * 2);
    __bf16* rmsgT = (__bf16*)alloc((size_t)RR * HH * DK * DK * 2);
    int* tcnt = (int*)alloc(TT * 4);
    int* rcnt = (int*)alloc(RR * 4);
    int* toff = (int*)alloc(TT * 4);
    int* roff = (int*)alloc(RR * 4);
    int* tcur = (int*)alloc(TT * 4);
    int* rcur = (int*)alloc(RR * 4);
    int* nsorted = (int*)alloc((size_t)NN * 4);
    int* esorted = (int*)alloc((size_t)EE * 4);
    int* deg     = (int*)alloc((size_t)NN * 4);
    float*  qn    = (float*)alloc((size_t)NN * DD * 4);
    __bf16* knbf  = (__bf16*)alloc((size_t)NN * DD * 2);
    __bf16* vnbf  = (__bf16*)alloc((size_t)NN * DD * 2);
    float*    lg   = (float*)alloc((size_t)EE * HH * 4);
    unsigned* mxu  = (unsigned*)alloc((size_t)NN * HH * 4);
    float*    ssum = (float*)alloc((size_t)NN * HH * 4);
    float*  vps    = (float*)alloc((size_t)NN * DD * 4);
    __bf16* aggbf  = (__bf16*)alloc((size_t)KM * NN * DD * 2);
    float*  aggm   = (float*)alloc((size_t)KM * NN * DD * 4);
    __bf16* aggmbf = (__bf16*)alloc((size_t)KM * NN * DD * 2);
    __bf16* qlinbf = (__bf16*)alloc((size_t)NN * DD * 2);
    float*  front  = (float*)alloc((size_t)NN * DD * 4);
    float*  tail   = (float*)alloc((size_t)NN * DD * 4);
    float*  res    = (float*)alloc((size_t)NN * DD * 4);
    __bf16* hbf    = (__bf16*)alloc((size_t)NN * DD * 2);
    float*  trans  = (float*)alloc((size_t)NN * DD * 4);

    // zero-init (graph-capturable)
    hipMemsetAsync(tcnt, 0, TT * 4, stream);
    hipMemsetAsync(rcnt, 0, RR * 4, stream);
    hipMemsetAsync(deg,  0, (size_t)NN * 4, stream);
    hipMemsetAsync(mxu,  0, (size_t)NN * HH * 4, stream);   // 0 < every enc()
    hipMemsetAsync(ssum, 0, (size_t)NN * HH * 4, stream);
    hipMemsetAsync(res,  0, (size_t)NN * DD * 4, stream);

    const int TB = 256;
    count_kernel<<<(EE + TB - 1) / TB, TB, 0, stream>>>(ntype, etype, tgt, tcnt, rcnt, deg);
    scan_kernel<<<1, 1, 0, stream>>>(tcnt, rcnt, toff, roff, tcur, rcur);
    scatter_kernel<<<(EE + TB - 1) / TB, TB, 0, stream>>>(ntype, etype, tcur, rcur, nsorted, esorted);

    cast_x_kernel<<<(NN * DD + TB - 1) / TB, TB, 0, stream>>>(meta_xs, xbf, NN * DD);
    auto ct = [&](const float* s, __bf16* d, int dim, int nm) {
        int tot = nm * dim * dim;
        castT_kernel<<<(tot + TB - 1) / TB, TB, 0, stream>>>(s, d, dim, nm);
    };
    ct(q_w, qwT, DD, TT);  ct(k_w, kwT, DD, TT);  ct(v_w, vwT, DD, TT);  ct(a_w, awT, DD, TT);
    ct(WMk, WMkT, DD, KM); ct(Wak, WakT, DD, KM); ct(Wq, WqT, DD, 1);    ct(Wkl, WklT, DD, 1);
    ct(rel_att, rattT, DK, RR * HH);  ct(rel_msg, rmsgT, DK, RR * HH);

    // typed q/k/v projections (WMMA over type-sorted tiles)
    typed_qkv_kernel<<<NN / 16 + TT, 256, 0, stream>>>(
        xbf, nsorted, tcnt, toff, qwT, kwT, vwT, q_b, k_b, v_b, qn, knbf, vnbf);

    // edge stage: relation-sorted WMMA + logits + segment max
    edge_attn_kernel<<<EE / 16 + RR, 256, 0, stream>>>(
        knbf, vnbf, qn, esorted, rcnt, roff, src, tgt, rattT, rmsgT, rel_pri,
        lg, mxu, vps);

    softmax_ex_kernel<<<(EE * HH + TB - 1) / TB, TB, 0, stream>>>(tgt, lg, mxu, ssum);
    moments_kernel<<<(NN * DD + TB - 1) / TB, TB, 0, stream>>>(vps, lg, ssum, deg, tgt, aggbf);

    // aggm[k] = aggmod[k] @ WMk[k]
    for (int kk = 0; kk < KM; ++kk)
        gemm128_kernel<<<(NN + 15) / 16, 256, 0, stream>>>(
            aggbf + (size_t)kk * NN * DD, WMkT + (size_t)kk * DD * DD, nullptr,
            aggm + (size_t)kk * NN * DD, aggmbf + (size_t)kk * NN * DD, NN);

    // qlin = x @ Wq + bq (bf16 copy only)
    gemm128_kernel<<<(NN + 15) / 16, 256, 0, stream>>>(
        xbf, WqT, bq, nullptr, qlinbf, NN);

    for (int kk = 0; kk < KM; ++kk) {
        gemm128_kernel<<<(NN + 15) / 16, 256, 0, stream>>>(
            qlinbf, WakT + (size_t)kk * DD * DD, nullptr, front, nullptr, NN);
        gemm128_kernel<<<(NN + 15) / 16, 256, 0, stream>>>(
            aggmbf + (size_t)kk * NN * DD, WklT, bkl, tail, nullptr, NN);
        gate_kernel<<<NN, DD, 0, stream>>>(front, tail, aggm + (size_t)kk * NN * DD, res);
    }

    gelu_cast_kernel<<<(NN * DD + TB - 1) / TB, TB, 0, stream>>>(res, hbf, NN * DD);
    typed_out_kernel<<<NN / 16 + TT, 256, 0, stream>>>(
        hbf, nsorted, tcnt, toff, awT, a_b, trans);
    final_ln_kernel<<<NN, DD, 0, stream>>>(trans, meta_xs, ntype, skip, ln_g, ln_b, out);
}
